// Attention_25933012533683
// MI455X (gfx1250) — matrix operand order repaired and checked by reading, not compiled, not run on hardware
//
#include <hip/hip_runtime.h>

// ---------------------------------------------------------------------------
// CDNA5 (gfx1250) attention layer: QKV proj + causal flash attention + proj.
// All matmuls via v_wmma_f32_16x16x32_bf16 (wave32).
// LDS tiles stored so every fragment read is a pair of 128-bit ds loads;
// all f32->bf16 conversion through native packed converts.
// ---------------------------------------------------------------------------

typedef __attribute__((ext_vector_type(16))) __bf16 v16bf;
typedef __attribute__((ext_vector_type(8)))  __bf16 v8bf;
typedef __attribute__((ext_vector_type(2)))  __bf16 v2bf;
typedef __attribute__((ext_vector_type(8)))  float  v8f;
typedef __attribute__((ext_vector_type(2)))  float  v2f;

#define S_LEN 2048
#define D_DIM 1024
#define N_HEAD 16
#define HEAD_D 64

// native scalar convert (hardware bf16 cvt)
__device__ __forceinline__ __bf16 f2bf(float f) { return (__bf16)f; }

// pack two floats as bf16 pair in a dword (lo in [15:0]) -> v_cvt_pk_bf16_f32
__device__ __forceinline__ unsigned pk2(float lo, float hi) {
    v2f f = {lo, hi};
    v2bf b = __builtin_convertvector(f, v2bf);
    union { v2bf b; unsigned u; } o; o.b = b;
    return o.u;
}

// convert 8 floats -> 8 bf16 (packed converts)
__device__ __forceinline__ v8bf cvt8(v8f f) {
    return __builtin_convertvector(f, v8bf);
}

// A-matrix fragment (16x32 bf16) from a row-major LDS tile (ldk mult of 8).
// ISA 7.12.2: lane L (half=L>>4, r=L&15) holds row r;
//   elems 0..7  -> K = k0 + half*8 + j ;  elems 8..15 -> K = k0+16+half*8+j
__device__ __forceinline__ v16bf load_A_frag(const __bf16* tile, int ldk,
                                             int row0, int k0) {
    int lane = threadIdx.x & 31;
    int r = lane & 15, half = lane >> 4;
    const __bf16* p = tile + (size_t)(row0 + r) * ldk + k0 + half * 8;
    v8bf lo = *(const v8bf*)p;
    v8bf hi = *(const v8bf*)(p + 16);
    return __builtin_shufflevector(lo, hi, 0, 1, 2, 3, 4, 5, 6, 7,
                                           8, 9, 10, 11, 12, 13, 14, 15);
}

// B-matrix fragment (32x16) from an LDS tile stored K-CONTIGUOUS: tile[n][k],
// ldk mult of 8.  elems 0..7 -> K = half*8 + j ; elems 8..15 -> K = 16+half*8+j.
__device__ __forceinline__ v16bf load_B_fragT(const __bf16* tile, int ldk,
                                              int k0, int col0) {
    int lane = threadIdx.x & 31;
    int c = lane & 15, half = lane >> 4;
    const __bf16* p = tile + (size_t)(col0 + c) * ldk + k0 + half * 8;
    v8bf lo = *(const v8bf*)p;
    v8bf hi = *(const v8bf*)(p + 16);
    return __builtin_shufflevector(lo, hi, 0, 1, 2, 3, 4, 5, 6, 7,
                                           8, 9, 10, 11, 12, 13, 14, 15);
}

__device__ __forceinline__ v8f wmma_bf16(v16bf a, v16bf b, v8f c) {
    return __builtin_amdgcn_wmma_f32_16x16x32_bf16(
        false, a, false, b, (short)0, c, false, false);
}

// ---------------------------------------------------------------------------
// GEMM: out = X[M,K] @ W[K,N] + bias.  Block tile 128x64, 256 thr = 8 waves,
// wave w -> 16 rows x 64 cols.  kv_c < 0: row-major out.
// kv_c in {0,1}: scatter into cached_kv layout [B,2,H,S,HD].
// ---------------------------------------------------------------------------
__global__ __launch_bounds__(256)
void gemm_bias_kernel(const float* __restrict__ X, const float* __restrict__ W,
                      const float* __restrict__ bias, float* __restrict__ out,
                      int M, int N, int K, int kv_c) {
    __shared__ __align__(16) __bf16 sX[128][32];   // [m][k]
    __shared__ __align__(16) __bf16 sWt[64][40];   // [n][k] (k-contiguous)

    const int n0 = blockIdx.x * 64;
    const int m0 = blockIdx.y * 128;
    const int tid = threadIdx.x;
    const int w = tid >> 5;
    const int lane = tid & 31;
    const int half = lane >> 4, c16 = lane & 15;

    const v8f vzero = {0.f, 0.f, 0.f, 0.f, 0.f, 0.f, 0.f, 0.f};
    v8f acc[4];
#pragma unroll
    for (int i = 0; i < 4; ++i) acc[i] = vzero;

    // staging coordinates (constant per thread)
    const int xrow = tid >> 1, xcb = (tid & 1) * 16;        // X: 16 elems
    const int wkp = tid >> 4, wnb = (tid & 15) * 4;         // W: 2k x 4n

    for (int k0 = 0; k0 < K; k0 += 32) {
        // ---- stage X tile (128x32) f32 -> bf16, vector converts ----
        {
            const float* src = X + (size_t)(m0 + xrow) * K + k0 + xcb;
            v8f f0 = *(const v8f*)(src);
            v8f f1 = *(const v8f*)(src + 8);
            *(v8bf*)&sX[xrow][xcb]     = cvt8(f0);
            *(v8bf*)&sX[xrow][xcb + 8] = cvt8(f1);
        }
        // ---- stage W tile (32x64) transposed -> sWt[n][k] ----
        {
            const float* r0 = W + (size_t)(k0 + 2 * wkp) * N + n0 + wnb;
            const float* r1 = r0 + N;
            float4 a = *(const float4*)r0;
            float4 b = *(const float4*)r1;
            *(unsigned*)&sWt[wnb + 0][2 * wkp] = pk2(a.x, b.x);
            *(unsigned*)&sWt[wnb + 1][2 * wkp] = pk2(a.y, b.y);
            *(unsigned*)&sWt[wnb + 2][2 * wkp] = pk2(a.z, b.z);
            *(unsigned*)&sWt[wnb + 3][2 * wkp] = pk2(a.w, b.w);
        }
        if (k0 + 32 < K) {
            __builtin_prefetch(X + (size_t)(m0 + xrow) * K + k0 + 32 + xcb, 0, 1);
            __builtin_prefetch(W + (size_t)(k0 + 32 + 2 * wkp) * N + n0 + wnb, 0, 1);
        }
        __syncthreads();

        v16bf a = load_A_frag(&sX[0][0], 32, w * 16, 0);
#pragma unroll
        for (int t = 0; t < 4; ++t) {
            v16bf b = load_B_fragT(&sWt[0][0], 40, 0, t * 16);
            acc[t] = wmma_bf16(a, b, acc[t]);
        }
        __syncthreads();
    }

    // epilogue: C layout -> global
#pragma unroll
    for (int t = 0; t < 4; ++t) {
#pragma unroll
        for (int r = 0; r < 8; ++r) {
            int row = m0 + w * 16 + half * 8 + r;
            int col = n0 + t * 16 + c16;
            float v = acc[t][r] + bias[col];
            if (kv_c < 0) {
                out[(size_t)row * N + col] = v;
            } else {
                int b = row >> 11, si = row & 2047;
                int h = col >> 6,  hd = col & 63;
                out[((((size_t)b * 2 + kv_c) * N_HEAD + h) * S_LEN + si)
                    * HEAD_D + hd] = v;
            }
        }
    }
}

// ---------------------------------------------------------------------------
// Causal flash attention.  grid = (S/64, B*H), block = 128 (4 waves).
// Wave w owns 16 query rows; keys streamed 32 at a time via LDS (bf16).
// ---------------------------------------------------------------------------
__global__ __launch_bounds__(128)
void attn_kernel(const float* __restrict__ Q,   // [B*S, D] row-major
                 const float* __restrict__ KV,  // [B,2,H,S,HD]
                 float* __restrict__ Oout) {    // [B*S, D] row-major
    __shared__ __align__(16) __bf16 sK[32][72];      // [key][hd]
    __shared__ __align__(16) __bf16 sVt[64][40];     // [hd][key] (k-contig)
    __shared__ __align__(16) __bf16 sP[4][16][32];   // per-wave P staging

    const int bh = blockIdx.y;
    const int b = bh / N_HEAD, h = bh % N_HEAD;
    const int qb = blockIdx.x * 64;
    const int tid = threadIdx.x;
    const int w = tid >> 5;
    const int lane = tid & 31;
    const int half = lane >> 4, c16 = lane & 15;

    const float* Kb = KV + (((size_t)b * 2 + 0) * N_HEAD + h) * S_LEN * HEAD_D;
    const float* Vb = KV + (((size_t)b * 2 + 1) * N_HEAD + h) * S_LEN * HEAD_D;

    // preload this wave's Q fragments (16 rows x 64, two K-chunks of 32)
    v16bf aq[2];
    {
        const float* qp =
            Q + (size_t)(b * S_LEN + qb + w * 16 + c16) * D_DIM + h * HEAD_D;
#pragma unroll
        for (int f = 0; f < 2; ++f) {
            int base = f * 32 + half * 8;
            v8f lo = *(const v8f*)(qp + base);
            v8f hi = *(const v8f*)(qp + base + 16);
            aq[f] = __builtin_shufflevector(cvt8(lo), cvt8(hi),
                                            0, 1, 2, 3, 4, 5, 6, 7,
                                            8, 9, 10, 11, 12, 13, 14, 15);
        }
    }

    const v8f vzero = {0.f, 0.f, 0.f, 0.f, 0.f, 0.f, 0.f, 0.f};
    v8f o[4];
#pragma unroll
    for (int t = 0; t < 4; ++t) o[t] = vzero;
    float m_r[8], l_r[8];
#pragma unroll
    for (int r = 0; r < 8; ++r) { m_r[r] = -1e30f; l_r[r] = 0.f; }

    // staging coordinates
    const int kkey = tid >> 2, kcb = (tid & 3) * 16;   // K: 16 elems, 1 key
    const int vkp = tid >> 3, vhb = (tid & 7) * 8;     // V: 2 keys x 8 hd

    const int nkt = (qb + 64) >> 5;   // uniform over the block
    for (int kt = 0; kt < nkt; ++kt) {
        // ---- stage K tile (32 keys x 64 hd), row-major, vector converts ----
        {
            const float* ks = Kb + (size_t)(kt * 32 + kkey) * HEAD_D + kcb;
            v8f f0 = *(const v8f*)(ks);
            v8f f1 = *(const v8f*)(ks + 8);
            *(v8bf*)&sK[kkey][kcb]     = cvt8(f0);
            *(v8bf*)&sK[kkey][kcb + 8] = cvt8(f1);
        }
        // ---- stage V tile transposed -> sVt[hd][key] ----
        {
            const float* r0 = Vb + (size_t)(kt * 32 + 2 * vkp) * HEAD_D + vhb;
            const float* r1 = r0 + HEAD_D;
            float4 a0 = *(const float4*)r0;
            float4 a1 = *(const float4*)(r0 + 4);
            float4 b0 = *(const float4*)r1;
            float4 b1 = *(const float4*)(r1 + 4);
            *(unsigned*)&sVt[vhb + 0][2 * vkp] = pk2(a0.x, b0.x);
            *(unsigned*)&sVt[vhb + 1][2 * vkp] = pk2(a0.y, b0.y);
            *(unsigned*)&sVt[vhb + 2][2 * vkp] = pk2(a0.z, b0.z);
            *(unsigned*)&sVt[vhb + 3][2 * vkp] = pk2(a0.w, b0.w);
            *(unsigned*)&sVt[vhb + 4][2 * vkp] = pk2(a1.x, b1.x);
            *(unsigned*)&sVt[vhb + 5][2 * vkp] = pk2(a1.y, b1.y);
            *(unsigned*)&sVt[vhb + 6][2 * vkp] = pk2(a1.z, b1.z);
            *(unsigned*)&sVt[vhb + 7][2 * vkp] = pk2(a1.w, b1.w);
        }
        if (kt + 1 < nkt) {
            __builtin_prefetch(Kb + (size_t)((kt + 1) * 32 + kkey) * HEAD_D + kcb, 0, 1);
            __builtin_prefetch(Vb + (size_t)((kt + 1) * 32 + 2 * vkp) * HEAD_D + vhb, 0, 1);
        }
        __syncthreads();

        // scores: S(16x32) = Q(16x64) @ K_tile^T   (4 WMMAs)
        v8f s0 = vzero, s1 = vzero;
        s0 = wmma_bf16(aq[0], load_B_fragT(&sK[0][0], 72, 0, 0), s0);
        s0 = wmma_bf16(aq[1], load_B_fragT(&sK[0][0], 72, 32, 0), s0);
        s1 = wmma_bf16(aq[0], load_B_fragT(&sK[0][0], 72, 0, 16), s1);
        s1 = wmma_bf16(aq[1], load_B_fragT(&sK[0][0], 72, 32, 16), s1);

        // scale + causal mask + online softmax
#pragma unroll
        for (int r = 0; r < 8; ++r) {
            int qrow = qb + w * 16 + half * 8 + r;
            int key0 = kt * 32 + c16;
            float a0 = s0[r] * 0.125f, a1 = s1[r] * 0.125f;
            if (key0 > qrow)       a0 = -1e30f;
            if (key0 + 16 > qrow)  a1 = -1e30f;

            float mx = fmaxf(a0, a1);
            mx = fmaxf(mx, __shfl_xor(mx, 8, 16));
            mx = fmaxf(mx, __shfl_xor(mx, 4, 16));
            mx = fmaxf(mx, __shfl_xor(mx, 2, 16));
            mx = fmaxf(mx, __shfl_xor(mx, 1, 16));

            float m_new = fmaxf(m_r[r], mx);
            float alpha = __expf(m_r[r] - m_new);
            float p0 = __expf(a0 - m_new);
            float p1 = __expf(a1 - m_new);
            float rs = p0 + p1;
            rs += __shfl_xor(rs, 8, 16);
            rs += __shfl_xor(rs, 4, 16);
            rs += __shfl_xor(rs, 2, 16);
            rs += __shfl_xor(rs, 1, 16);

            l_r[r] = l_r[r] * alpha + rs;
            m_r[r] = m_new;
            s0[r] = p0; s1[r] = p1;
#pragma unroll
            for (int t = 0; t < 4; ++t) o[t][r] *= alpha;
        }

        // C-layout -> A-layout transpose of P through per-wave LDS staging
#pragma unroll
        for (int r = 0; r < 8; ++r) {
            sP[w][half * 8 + r][c16]      = f2bf(s0[r]);
            sP[w][half * 8 + r][16 + c16] = f2bf(s1[r]);
        }
        // wave-private LDS round trip: drain DS counter before reading back
        asm volatile("s_wait_dscnt 0" ::: "memory");

        v16bf ap = load_A_frag(&sP[w][0][0], 32, 0, 0);
#pragma unroll
        for (int t = 0; t < 4; ++t)
            o[t] = wmma_bf16(ap, load_B_fragT(&sVt[0][0], 40, 0, t * 16), o[t]);

        __syncthreads();
    }

    // normalize and write attention output (row-major [B*S, D])
    float inv_l[8];
#pragma unroll
    for (int r = 0; r < 8; ++r) inv_l[r] = 1.0f / l_r[r];
#pragma unroll
    for (int t = 0; t < 4; ++t) {
#pragma unroll
        for (int r = 0; r < 8; ++r) {
            int row = b * S_LEN + qb + w * 16 + half * 8 + r;
            int col = h * HEAD_D + t * 16 + c16;
            Oout[(size_t)row * D_DIM + col] = o[t][r] * inv_l[r];
        }
    }
}

// ---------------------------------------------------------------------------
extern "C" void kernel_launch(void* const* d_in, const int* in_sizes, int n_in,
                              void* d_out, int out_size, void* d_ws,
                              size_t ws_size, hipStream_t stream) {
    (void)in_sizes; (void)n_in; (void)out_size; (void)ws_size;

    const float* x  = (const float*)d_in[0];
    const float* Wq = (const float*)d_in[1];
    const float* bq = (const float*)d_in[2];
    const float* Wk = (const float*)d_in[3];
    const float* bk = (const float*)d_in[4];
    const float* Wv = (const float*)d_in[5];
    const float* bv = (const float*)d_in[6];
    const float* Wp = (const float*)d_in[7];
    const float* bp = (const float*)d_in[8];

    const int B = 2, M = B * S_LEN;                   // 4096
    float* y  = (float*)d_out;                        // [B,S,D]
    float* kv = (float*)d_out + (size_t)M * D_DIM;    // [B,2,H,S,HD]

    float* q_scr    = (float*)d_ws;                       // 16 MB
    float* attn_scr = q_scr + (size_t)M * D_DIM;          // 16 MB

    dim3 ggrid(D_DIM / 64, M / 128);   // (16, 32)
    dim3 gblk(256);

    // QKV projections (K/V straight into cached_kv layout)
    gemm_bias_kernel<<<ggrid, gblk, 0, stream>>>(x, Wq, bq, q_scr, M, D_DIM, D_DIM, -1);
    gemm_bias_kernel<<<ggrid, gblk, 0, stream>>>(x, Wk, bk, kv,    M, D_DIM, D_DIM, 0);
    gemm_bias_kernel<<<ggrid, gblk, 0, stream>>>(x, Wv, bv, kv,    M, D_DIM, D_DIM, 1);

    // causal flash attention
    dim3 agrid(S_LEN / 64, B * N_HEAD);   // (32, 32)
    attn_kernel<<<agrid, dim3(128), 0, stream>>>(q_scr, kv, attn_scr);

    // output projection
    gemm_bias_kernel<<<ggrid, gblk, 0, stream>>>(attn_scr, Wp, bp, y, M, D_DIM, D_DIM, -1);
}
